// HeteroGraphConv_17532056502698
// MI455X (gfx1250) — matrix operand order repaired and checked by reading, not compiled, not run on hardware
//
#include <hip/hip_runtime.h>
#include <hip/hip_bf16.h>

typedef __attribute__((ext_vector_type(16))) __bf16 v16bf;
typedef __attribute__((ext_vector_type(8)))  float  v8f;

#define N_NODES 50000
#define E_EDGES 800000
#define DIN     64
#define DEF     16
#define MSGW    32
#define HID     16
#define DOUT    64
#define WAVES   8   // 256-thread block = 8 wave32

// ---------- helpers ----------

__device__ __forceinline__ v8f wmma_bf16(v16bf a, v16bf b, v8f c) {
    // D = A(16x32 bf16) * B(32x16 bf16) + C(16x16 f32)
    return __builtin_amdgcn_wmma_f32_16x16x32_bf16(false, a, false, b,
                                                   (short)0, c, false, false);
}

// Monotone order-preserving float -> u32 mapping so scatter-max can use the
// native non-returning GLOBAL_ATOMIC_MAX_U32 instead of a float CAS loop.
__device__ __forceinline__ unsigned fmap(float f) {
    unsigned u = __float_as_uint(f);
    return (u & 0x80000000u) ? ~u : (u | 0x80000000u);
}
__device__ __forceinline__ float funmap(unsigned u) {
    unsigned b = (u & 0x80000000u) ? (u & 0x7FFFFFFFu) : ~u;
    return __uint_as_float(b);
}
// fmap(-inf) == 0x007FFFFF : init value for the reduce buffers
#define FMAP_NEG_INF 0x007FFFFFu

// Low 32 bits of a generic pointer to __shared__ == LDS byte address
__device__ __forceinline__ unsigned lds_addr_of(const void* p) {
    return (unsigned)(unsigned long long)p;
}

__global__ __launch_bounds__(256) void init_reduce_kernel(unsigned* __restrict__ p, int n) {
    int i = blockIdx.x * 256 + threadIdx.x;
    if (i < n) p[i] = FMAP_NEG_INF;
}

// ---------- edge kernel: m = MLP2([ef | x_src[src]]) ; scatter-max to r ----------
// One wave per 16-edge tile; double-buffered async global->LDS staging.
__global__ __launch_bounds__(256) void edge_msg_kernel(
    const float* __restrict__ xsrc,   // N x 64
    const int*   __restrict__ src,    // E
    const int*   __restrict__ dst,    // E
    const float* __restrict__ ef,     // E x 16
    const float* __restrict__ W1,     // 80 x 16 (row-major [fi][fo])
    const float* __restrict__ b1,     // 16
    const float* __restrict__ W2,     // 16 x 32
    const float* __restrict__ b2,     // 32
    unsigned*    __restrict__ r,      // N x 32 (fmap-encoded), pre-init fmap(-inf)
    int n_tiles)
{
    // f32 staging tiles, double buffered, filled by async global->LDS copies
    __shared__ float  sXf[WAVES][2][16][96];  // cols: 0..15 ef | 16..79 x | 80..95 zero pad
    __shared__ __bf16 sH[WAVES][16][16];      // hidden-layer transpose scratch

    const int tid  = threadIdx.x;
    const int w    = tid >> 5;
    const int lane = tid & 31;
    const int half = lane >> 4;   // 0: lanes 0-15, 1: lanes 16-31
    const int m    = lane & 15;
    const int row  = lane >> 1;   // staging: 2 lanes per edge row
    const int h    = lane & 1;

    // ---- loop-invariant weight fragments (B operands), built once per wave ----
    // B(32x16) layout: element i -> K = 16*half + i, N = lane&15
    v16bf bw1[3];
#pragma unroll
    for (int c = 0; c < 3; ++c)
#pragma unroll
        for (int i = 0; i < 16; ++i) {
            int K = c * 32 + (half << 4) + i;
            bw1[c][i] = (K < 80) ? (__bf16)W1[K * 16 + m] : (__bf16)0.f;
        }
    v16bf bw2[2];
#pragma unroll
    for (int h2 = 0; h2 < 2; ++h2)
#pragma unroll
        for (int i = 0; i < 16; ++i) {
            int K = (half << 4) + i;
            bw2[h2][i] = (K < 16) ? (__bf16)W2[K * 32 + h2 * 16 + m] : (__bf16)0.f;
        }
    const float bias1   = b1[m];
    const float bias2lo = b2[m];
    const float bias2hi = b2[16 + m];

    // zero the K-padding columns (80..95) in both buffers once; async copies
    // only ever write cols 0..79
#pragma unroll
    for (int b = 0; b < 2; ++b)
#pragma unroll
        for (int c2 = 0; c2 < 8; ++c2) sXf[w][b][row][80 + h * 8 + c2] = 0.f;

    // per-lane LDS destinations for the async copies (both buffers)
    unsigned ldsEf[2], ldsX[2];
#pragma unroll
    for (int b = 0; b < 2; ++b) {
        ldsEf[b] = lds_addr_of(&sXf[w][b][row][h * 8]);
        ldsX[b]  = lds_addr_of(&sXf[w][b][row][16 + h * 32]);
    }

    const int stride = gridDim.x * WAVES;
    const int t0 = blockIdx.x * WAVES + w;

    // issue one tile's staging: 32B of ef + 128B of gathered x per lane
#define ISSUE_ASYNC(tile_, b_, sidx_)                                               \
    do {                                                                            \
        long e_ = ((long)(tile_) << 4) + row;                                       \
        const float* efp_ = ef + e_ * 16 + h * 8;                                   \
        const float* xp_  = xsrc + (long)(sidx_) * 64 + h * 32;                     \
        asm volatile(                                                               \
            "global_load_async_to_lds_b128 %0, %2, off\n\t"                         \
            "global_load_async_to_lds_b128 %0, %2, off offset:16\n\t"               \
            "global_load_async_to_lds_b128 %1, %3, off\n\t"                         \
            "global_load_async_to_lds_b128 %1, %3, off offset:16\n\t"               \
            "global_load_async_to_lds_b128 %1, %3, off offset:32\n\t"               \
            "global_load_async_to_lds_b128 %1, %3, off offset:48\n\t"               \
            "global_load_async_to_lds_b128 %1, %3, off offset:64\n\t"               \
            "global_load_async_to_lds_b128 %1, %3, off offset:80\n\t"               \
            "global_load_async_to_lds_b128 %1, %3, off offset:96\n\t"               \
            "global_load_async_to_lds_b128 %1, %3, off offset:112"                  \
            :: "v"(ldsEf[b_]), "v"(ldsX[b_]), "v"(efp_), "v"(xp_) : "memory");      \
    } while (0)

    // ---- 2-stage pipeline: indices one stage ahead of async copies ----
    int sidx_n1 = 0, sidx_n2 = 0;
    if (t0 < n_tiles)          sidx_n1 = src[((long)t0 << 4) + row];
    if (t0 + stride < n_tiles) sidx_n2 = src[(((long)t0 + stride) << 4) + row];
    if (t0 < n_tiles) ISSUE_ASYNC(t0, 0, sidx_n1);

    int buf = 0;
    for (int tile = t0; tile < n_tiles; tile += stride) {
        // current buffer's async copy complete
        asm volatile("s_wait_asynccnt 0x0" ::: "memory");

        // kick off next tile's copy; it overlaps this tile's compute + scatter
        int t_next = tile + stride;
        if (t_next < n_tiles) {
            ISSUE_ASYNC(t_next, buf ^ 1, sidx_n2);
            int t_nn = t_next + stride;
            if (t_nn < n_tiles) sidx_n2 = src[((long)t_nn << 4) + row];
        }

        // ---- H = relu(X @ W1 + b1) : three 16x16x32 WMMAs ----
        v8f hc;
#pragma unroll
        for (int j = 0; j < 8; ++j) hc[j] = bias1;
#pragma unroll
        for (int c = 0; c < 3; ++c) {
            v16bf a;
#pragma unroll
            for (int i = 0; i < 16; ++i) {
                int K = c * 32 + (half << 4) + i;
                a[i] = (__bf16)sXf[w][buf][m][K];
            }
            hc = wmma_bf16(a, bw1[c], hc);
        }

        // ---- ReLU + transpose (C-layout -> A-layout) through LDS ----
#pragma unroll
        for (int j = 0; j < 8; ++j) {
            float v = hc[j] > 0.f ? hc[j] : 0.f;
            sH[w][(half << 3) + j][m] = (__bf16)v;
        }
        asm volatile("s_wait_dscnt 0x0" ::: "memory");

        v16bf a2;
#pragma unroll
        for (int i = 0; i < 16; ++i) {
            int K = (half << 4) + i;
            a2[i] = (K < 16) ? sH[w][m][K] : (__bf16)0.f;
        }

        // ---- M = H @ W2 + b2 : 16 x 32 message tile ----
        v8f c0, c1;
#pragma unroll
        for (int j = 0; j < 8; ++j) { c0[j] = bias2lo; c1[j] = bias2hi; }
        c0 = wmma_bf16(a2, bw2[0], c0);
        c1 = wmma_bf16(a2, bw2[1], c1);

        // ---- scatter-max onto dst nodes (native u32 atomic max) ----
#pragma unroll
        for (int j = 0; j < 8; ++j) {
            int e = (tile << 4) + (half << 3) + j;
            int d = dst[e];
            atomicMax(&r[(long)d * 32 + m],      fmap(c0[j]));
            atomicMax(&r[(long)d * 32 + 16 + m], fmap(c1[j]));
        }
        buf ^= 1;
    }
#undef ISSUE_ASYNC
}

// ---------- node kernel: out = MLP2([x_dst | r]) (optionally max-combined) ----------
__global__ __launch_bounds__(256) void node_update_kernel(
    const float*    __restrict__ xdst,  // N x 64
    const unsigned* __restrict__ r,     // N x 32 (fmap-encoded)
    const float*    __restrict__ W1,    // 96 x 16
    const float*    __restrict__ b1,    // 16
    const float*    __restrict__ W2,    // 16 x 64
    const float*    __restrict__ b2,    // 64
    float*          __restrict__ out,   // N x 64
    int combine, int n_tiles)
{
    __shared__ __bf16 sX[WAVES][16][96];
    __shared__ __bf16 sH[WAVES][16][16];

    const int tid  = threadIdx.x;
    const int w    = tid >> 5;
    const int lane = tid & 31;
    const int half = lane >> 4;
    const int m    = lane & 15;

    v16bf bw1[3];
#pragma unroll
    for (int c = 0; c < 3; ++c)
#pragma unroll
        for (int i = 0; i < 16; ++i) {
            int K = c * 32 + (half << 4) + i;          // K < 96 always
            bw1[c][i] = (__bf16)W1[K * 16 + m];
        }
    v16bf bw2[4];
#pragma unroll
    for (int h2 = 0; h2 < 4; ++h2)
#pragma unroll
        for (int i = 0; i < 16; ++i) {
            int K = (half << 4) + i;
            bw2[h2][i] = (K < 16) ? (__bf16)W2[K * 64 + h2 * 16 + m] : (__bf16)0.f;
        }
    const float bias1 = b1[m];
    float bias2[4];
#pragma unroll
    for (int h2 = 0; h2 < 4; ++h2) bias2[h2] = b2[h2 * 16 + m];

    for (int tile = blockIdx.x * WAVES + w; tile < n_tiles; tile += gridDim.x * WAVES) {
        const int n0 = tile << 4;

        // ---- stage [x_dst | decoded r] ; 2 lanes per node row ----
        {
            int row = lane >> 1, h = lane & 1;
            int node = n0 + row;
            const float4* xr = (const float4*)(xdst + (long)node * 64 + h * 32);
#pragma unroll
            for (int c2 = 0; c2 < 8; ++c2) {
                float4 v = xr[c2];
                __bf16* p = &sX[w][row][h * 32 + c2 * 4];
                p[0] = (__bf16)v.x; p[1] = (__bf16)v.y;
                p[2] = (__bf16)v.z; p[3] = (__bf16)v.w;
            }
            const unsigned* rr = r + (long)node * 32 + h * 16;
#pragma unroll
            for (int c2 = 0; c2 < 16; ++c2) {
                float v = funmap(rr[c2]);
                // empty segments stay fmap(-inf); NaN fails the compare too
                if (!(v >= -3.0e38f)) v = 0.f;
                sX[w][row][64 + h * 16 + c2] = (__bf16)v;
            }
        }
        asm volatile("s_wait_dscnt 0x0" ::: "memory");

        // ---- H = relu(X @ W1 + b1) ----
        v8f hc;
#pragma unroll
        for (int j = 0; j < 8; ++j) hc[j] = bias1;
#pragma unroll
        for (int c = 0; c < 3; ++c) {
            v16bf a;
#pragma unroll
            for (int i = 0; i < 16; ++i) {
                int K = c * 32 + (half << 4) + i;
                a[i] = sX[w][m][K];
            }
            hc = wmma_bf16(a, bw1[c], hc);
        }
#pragma unroll
        for (int j = 0; j < 8; ++j) {
            float v = hc[j] > 0.f ? hc[j] : 0.f;
            sH[w][(half << 3) + j][m] = (__bf16)v;
        }
        asm volatile("s_wait_dscnt 0x0" ::: "memory");

        v16bf a2;
#pragma unroll
        for (int i = 0; i < 16; ++i) {
            int K = (half << 4) + i;
            a2[i] = (K < 16) ? sH[w][m][K] : (__bf16)0.f;
        }

        // ---- O = H @ W2 + b2 : 16 x 64 output tile (4 WMMAs) ----
#pragma unroll
        for (int h2 = 0; h2 < 4; ++h2) {
            v8f c;
#pragma unroll
            for (int j = 0; j < 8; ++j) c[j] = bias2[h2];
            c = wmma_bf16(a2, bw2[h2], c);
#pragma unroll
            for (int j = 0; j < 8; ++j) {
                int node = n0 + (half << 3) + j;
                long o = (long)node * 64 + h2 * 16 + m;
                float v = c[j];
                if (combine) v = fmaxf(v, out[o]);   // cross-relation max
                out[o] = v;
            }
        }
    }
}

// ---------- launch ----------
extern "C" void kernel_launch(void* const* d_in, const int* in_sizes, int n_in,
                              void* d_out, int out_size, void* d_ws, size_t ws_size,
                              hipStream_t stream)
{
    (void)in_sizes; (void)n_in; (void)out_size; (void)ws_size;

    const float* xA = (const float*)d_in[0];
    const float* xB = (const float*)d_in[1];
    const int*   src[4]; const int* dst[4]; const float* ef[4];
    for (int rel = 0; rel < 4; ++rel) {           // AA, AB, BA, BB
        src[rel] = (const int*)  d_in[2 + rel * 3];
        dst[rel] = (const int*)  d_in[3 + rel * 3];
        ef[rel]  = (const float*)d_in[4 + rel * 3];
    }
    // per-rel params: msgW1, msgb1, msgW2, msgb2, udtW1, udtb1, udtW2, udtb2
    const float* P[4][8];
    for (int rel = 0; rel < 4; ++rel)
        for (int k = 0; k < 8; ++k)
            P[rel][k] = (const float*)d_in[14 + rel * 8 + k];

    unsigned* rbuf[4];
    for (int rel = 0; rel < 4; ++rel)
        rbuf[rel] = (unsigned*)((char*)d_ws + (size_t)rel * N_NODES * 32 * sizeof(unsigned));

    // 1) init reduce buffers to fmap(-inf)
    int total = 4 * N_NODES * 32;
    init_reduce_kernel<<<(total + 255) / 256, 256, 0, stream>>>((unsigned*)d_ws, total);

    // 2) per-relation edge message + scatter-max
    const float* xs[4] = { xA, xA, xB, xB };      // src node type per relation
    const int n_etiles = E_EDGES / 16;            // 50000, exact
    for (int rel = 0; rel < 4; ++rel)
        edge_msg_kernel<<<1024, 256, 0, stream>>>(
            xs[rel], src[rel], dst[rel], ef[rel],
            P[rel][0], P[rel][1], P[rel][2], P[rel][3],
            rbuf[rel], n_etiles);

    // 3) node updates + cross-relation max
    float* outA = (float*)d_out;
    float* outB = outA + (size_t)N_NODES * DOUT;
    const int n_ntiles = N_NODES / 16;            // 3125, exact
    // dst type A <- relations AA (0) and BA (2)
    node_update_kernel<<<256, 256, 0, stream>>>(xA, rbuf[0], P[0][4], P[0][5], P[0][6], P[0][7], outA, 0, n_ntiles);
    node_update_kernel<<<256, 256, 0, stream>>>(xA, rbuf[2], P[2][4], P[2][5], P[2][6], P[2][7], outA, 1, n_ntiles);
    // dst type B <- relations AB (1), BB (3)
    node_update_kernel<<<256, 256, 0, stream>>>(xB, rbuf[1], P[1][4], P[1][5], P[1][6], P[1][7], outB, 0, n_ntiles);
    node_update_kernel<<<256, 256, 0, stream>>>(xB, rbuf[3], P[3][4], P[3][5], P[3][6], P[3][7], outB, 1, n_ntiles);
}
